// MyMoeEncoderLayer_72043781423418
// MI455X (gfx1250) — matrix operand order repaired and checked by reading, not compiled, not run on hardware
//
#include <hip/hip_runtime.h>
#include <hip/hip_bf16.h>
#include <math.h>

// Problem constants (fixed by the reference)
#define TKN   4096   // B*L tokens
#define DM    1024   // d_model
#define FFN_  4096
#define INTER_ 4096
#define NEXP  4
#define NHEAD 16
#define HDIM  64
#define BATCH 4
#define SEQ   1024
#define KCH   256    // K-chunk staged to LDS by the TDM (16 rows x 256 f32 = 16KB)

typedef __attribute__((ext_vector_type(16))) __bf16 v16bf;
typedef __attribute__((ext_vector_type(8)))  float  v8f;
typedef unsigned int u32x4 __attribute__((ext_vector_type(4)));
typedef int i32x4 __attribute__((ext_vector_type(4)));
typedef int i32x8 __attribute__((ext_vector_type(8)));

#define V8F_ZERO {0.f,0.f,0.f,0.f,0.f,0.f,0.f,0.f}

static __device__ __forceinline__ int lane_id() { return threadIdx.x & 31; }

// ---- WMMA fragment loaders (contiguous k = 16*hf + s, same order in A and B; wave32) ----

// A-matrix 16x32 bf16: lane l (r=l&15, hf=l>>4) holds row M=r,
// K = kbase + hf*16 .. +16 (contiguous, two 16B chunks)
static __device__ __forceinline__ v16bf load_a_bf16(const __bf16* base, int lda, int kbase) {
    int l = lane_id();
    int r = l & 15, hf = l >> 4;
    const __bf16* p = base + (size_t)r * lda + kbase + hf * 16;
    union { v16bf v; uint4 q[2]; } u;
    u.q[0] = *(const uint4*)(p);
    u.q[1] = *(const uint4*)(p + 8);
    return u.v;
}

// Same, but caller supplies this lane's own row pointer (gathered rows)
static __device__ __forceinline__ v16bf load_a_row(const __bf16* rowbase, int kbase) {
    int l = lane_id();
    int hf = l >> 4;
    const __bf16* p = rowbase + kbase + hf * 16;
    union { v16bf v; uint4 q[2]; } u;
    u.q[0] = *(const uint4*)(p);
    u.q[1] = *(const uint4*)(p + 8);
    return u.v;
}

// B-matrix 32x16 bf16 from W[N,K] row-major (B = W^T):
// lane l holds col N=l&15, K = kbase + (l>>4)*16 .. +16 (contiguous run in the W row)
static __device__ __forceinline__ v16bf load_b_bf16(const __bf16* W, int ldw, int kbase) {
    int l = lane_id();
    int n = l & 15, koff = kbase + ((l >> 4) << 4);
    const __bf16* p = W + (size_t)n * ldw + koff;
    union { v16bf v; uint4 q[2]; } u;
    u.q[0] = *(const uint4*)(p);
    u.q[1] = *(const uint4*)(p + 8);
    return u.v;
}

// B fragment from fp32 weights in global: 4 x b128 loads + in-register cvt
static __device__ __forceinline__ v16bf load_b_f32(const float* W, int ldw, int kbase) {
    int l = lane_id();
    int n = l & 15, koff = kbase + ((l >> 4) << 4);
    const float4* p = (const float4*)(W + (size_t)n * ldw + koff);
    v16bf f;
#pragma unroll
    for (int j = 0; j < 4; ++j) {
        float4 q = p[j];
        f[4 * j + 0] = (__bf16)q.x;
        f[4 * j + 1] = (__bf16)q.y;
        f[4 * j + 2] = (__bf16)q.z;
        f[4 * j + 3] = (__bf16)q.w;
    }
    return f;
}

// B fragment from an LDS-staged fp32 tile (16 rows x KCH), b128 DS loads + cvt
static __device__ __forceinline__ v16bf load_b_lds(const float* Wl, int kbase) {
    int l = lane_id();
    int n = l & 15, koff = kbase + ((l >> 4) << 4);
    const float4* p = (const float4*)(Wl + n * KCH + koff);
    v16bf f;
#pragma unroll
    for (int j = 0; j < 4; ++j) {
        float4 q = p[j];
        f[4 * j + 0] = (__bf16)q.x;
        f[4 * j + 1] = (__bf16)q.y;
        f[4 * j + 2] = (__bf16)q.z;
        f[4 * j + 3] = (__bf16)q.w;
    }
    return f;
}

static __device__ __forceinline__ v8f wmma_bf16(v16bf a, v16bf b, v8f c) {
    return __builtin_amdgcn_wmma_f32_16x16x32_bf16(false, a, false, b, (short)0, c, false, false);
}

static __device__ __forceinline__ float gelu_exact(float x) {
    return 0.5f * x * (1.0f + erff(x * 0.70710678118654752f));
}

// ---- Tensor Data Mover: async 2D fp32 tile (rows x tile_k) global -> LDS ----
// D# per CDNA5 ISA. 2D tensor: groups 2/3 unused (zeros).
static __device__ __forceinline__ void tdm_load_tile_f32(unsigned lds_off, const float* src,
                                                         int rows, int tile_k,
                                                         int row_stride_elems) {
    unsigned long long ga = (unsigned long long)(uintptr_t)src;
    u32x4 g0;
    g0[0] = 1u;                                                // count=1, user mode
    g0[1] = lds_off;                                           // lds_addr (bytes)
    g0[2] = (unsigned)ga;                                      // global_addr[31:0]
    g0[3] = (unsigned)((ga >> 32) & 0x01FFFFFFull) | (2u << 30); // addr[56:32] | type=2
    unsigned td0 = (unsigned)tile_k;                           // tensor_dim0 (elements)
    unsigned td1 = (unsigned)rows;                             // tensor_dim1
    unsigned long long st0 = (unsigned long long)row_stride_elems;
    i32x8 g1;
    g1[0] = (int)(2u << 16);                                   // data_size=2 (4B), mask=0
    g1[1] = (int)((td0 & 0xFFFFu) << 16);                      // td0[15:0] @ bits63:48
    g1[2] = (int)(((td0 >> 16) & 0xFFFFu) | ((td1 & 0xFFFFu) << 16));
    g1[3] = (int)(((td1 >> 16) & 0xFFFFu) | (((unsigned)tile_k & 0xFFFFu) << 16)); // tile_dim0
    g1[4] = (int)((unsigned)rows & 0xFFFFu);                   // tile_dim1; tile_dim2=0
    g1[5] = (int)(st0 & 0xFFFFFFFFull);                        // dim0 stride lo
    g1[6] = (int)((st0 >> 32) & 0xFFFFull);                    // dim0 stride hi
    g1[7] = 0;
    i32x4 z4 = {0, 0, 0, 0};
#if __clang_major__ >= 23
    i32x8 z8 = {0, 0, 0, 0, 0, 0, 0, 0};
    __builtin_amdgcn_tensor_load_to_lds(g0, g1, z4, z4, z8, 0);
#else
    __builtin_amdgcn_tensor_load_to_lds(g0, g1, z4, z4, 0);
#endif
}

// ---- fp32 -> bf16 convert (vectorized) ----
__global__ __launch_bounds__(256) void cvt_kernel(const float* __restrict__ in,
                                                  __bf16* __restrict__ out, int n4) {
    int i = blockIdx.x * 256 + threadIdx.x;
    if (i < n4) {
        float4 q = ((const float4*)in)[i];
        __bf16 o[4] = { (__bf16)q.x, (__bf16)q.y, (__bf16)q.z, (__bf16)q.w };
        *(uint2*)(out + (size_t)i * 4) = *(const uint2*)o;
    }
}

// ---- Dense WMMA GEMM with TDM-staged, double-buffered weight tiles ----
// Block = 4 waves sharing one 16-col N tile, covering 4 consecutive 16-row M tiles.
// grid.x = (M/64) * (N/16). Wave 0 drives the TDM; s_wait_tensorcnt + barrier publish.
// EPI: 0 f32 out | 1 bf16 out | 2 bf16 -> [B,NH,L,HD] * scale (Q/K)
//      3 bf16 -> [B,NH,HD,L] (V^T) | 4 gelu -> bf16 | 5 f32 out + residual
template <int EPI>
__global__ __launch_bounds__(128)
void gemm_bf16w(const __bf16* __restrict__ A, const float* __restrict__ W,
                const float* __restrict__ bias, const float* __restrict__ res,
                float* __restrict__ outF, __bf16* __restrict__ outB,
                int M, int N, int K, float scale) {
    __shared__ __align__(16) float Wlds[2][16 * KCH];
    int wave = threadIdx.x >> 5, lane = threadIdx.x & 31;
    int tilesN = N >> 4;
    int bm = blockIdx.x / tilesN, tn = blockIdx.x % tilesN;
    int tm = bm * 4 + wave;
    const __bf16* Ap = A + (size_t)tm * 16 * K;
    const float*  Wp = W + (size_t)tn * 16 * K;

    if (wave == 0) {
        tdm_load_tile_f32((unsigned)(uintptr_t)&Wlds[0][0], Wp, 16, KCH, K);
        __builtin_amdgcn_s_wait_tensorcnt(0);
    }
    __syncthreads();

    v8f acc = V8F_ZERO;
    int buf = 0;
    int nch = K / KCH;
    for (int c = 0; c < nch; ++c) {
        if (wave == 0 && c + 1 < nch)
            tdm_load_tile_f32((unsigned)(uintptr_t)&Wlds[buf ^ 1][0],
                              Wp + (size_t)(c + 1) * KCH, 16, KCH, K);
        int kg = c * KCH;
#pragma unroll
        for (int kk = 0; kk < KCH; kk += 32)
            acc = wmma_bf16(load_a_bf16(Ap, K, kg + kk), load_b_lds(&Wlds[buf][0], kk), acc);
        if (wave == 0) __builtin_amdgcn_s_wait_tensorcnt(0);
        __syncthreads();
        buf ^= 1;
    }

    int hf = lane >> 4, col = lane & 15;
    int gn = tn * 16 + col;
    float bs = bias ? bias[gn] : 0.f;
#pragma unroll
    for (int i = 0; i < 8; ++i) {
        int gm = tm * 16 + i + hf * 8;
        float v = acc[i] + bs;
        if (EPI == 0) {
            outF[(size_t)gm * N + gn] = v;
        } else if (EPI == 1) {
            outB[(size_t)gm * N + gn] = (__bf16)v;
        } else if (EPI == 2) {  // [B,NH,L,HD]
            int hh = gn >> 6, hd = gn & 63, bb = gm >> 10, ll = gm & 1023;
            outB[((((size_t)bb * NHEAD + hh) * SEQ + ll) << 6) + hd] = (__bf16)(v * scale);
        } else if (EPI == 3) {  // [B,NH,HD,L]
            int hh = gn >> 6, hd = gn & 63, bb = gm >> 10, ll = gm & 1023;
            outB[((((size_t)bb * NHEAD + hh) * HDIM + hd) << 10) + ll] = (__bf16)v;
        } else if (EPI == 4) {
            outB[(size_t)gm * N + gn] = (__bf16)gelu_exact(v);
        } else {  // 5: residual add -> f32
            outF[(size_t)gm * N + gn] = v + res[(size_t)gm * N + gn];
        }
    }
}

// ---- Flash-style attention: one wave per (b, h, 16-query tile) ----
__global__ __launch_bounds__(128)
void attn_kernel(const __bf16* __restrict__ Q,   // [B,NH,L,HD], pre-scaled
                 const __bf16* __restrict__ Kt,  // [B,NH,L,HD]
                 const __bf16* __restrict__ Vt,  // [B,NH,HD,L]
                 const float* __restrict__ mask, // [B,1,L,L]
                 const float* __restrict__ hmask,// [NH]
                 __bf16* __restrict__ O)         // [B,L,D]
{
    __shared__ __align__(16) __bf16 Pl[4][16 * 32];
    int wave = threadIdx.x >> 5, lane = threadIdx.x & 31;
    int id = blockIdx.x * 4 + wave;          // 0..4095
    int qt = id & 63, bh = id >> 6;
    int h = bh & (NHEAD - 1), b = bh >> 4;
    int hf = lane >> 4, col = lane & 15;

    const __bf16* qp = Q + ((size_t)bh * SEQ + (size_t)qt * 16) * HDIM;
    v16bf aq0 = load_a_bf16(qp, HDIM, 0);
    v16bf aq1 = load_a_bf16(qp, HDIM, 32);

    float mrow[8], srow[8];
#pragma unroll
    for (int i = 0; i < 8; ++i) { mrow[i] = -1e30f; srow[i] = 0.f; }
    v8f o0 = V8F_ZERO, o1 = V8F_ZERO, o2 = V8F_ZERO, o3 = V8F_ZERO;

    __bf16* pl = &Pl[wave][0];
    const __bf16* vbase = Vt + (size_t)bh * HDIM * SEQ;
    const float*  mbase = mask + (size_t)b * SEQ * SEQ;

    for (int kp = 0; kp < SEQ / 32; ++kp) {
        int k0 = kp * 32;
        const __bf16* kr0 = Kt + ((size_t)bh * SEQ + k0) * HDIM;
        const __bf16* kr1 = kr0 + 16 * HDIM;
        v8f s0 = V8F_ZERO, s1 = V8F_ZERO;
        s0 = wmma_bf16(aq0, load_b_bf16(kr0, HDIM, 0), s0);
        s0 = wmma_bf16(aq1, load_b_bf16(kr0, HDIM, 32), s0);
        s1 = wmma_bf16(aq0, load_b_bf16(kr1, HDIM, 0), s1);
        s1 = wmma_bf16(aq1, load_b_bf16(kr1, HDIM, 32), s1);

        float f[8];
#pragma unroll
        for (int i = 0; i < 8; ++i) {
            int qrow = qt * 16 + i + hf * 8;
            float e0 = s0[i] + mbase[(size_t)qrow * SEQ + k0 + col];
            float e1 = s1[i] + mbase[(size_t)qrow * SEQ + k0 + 16 + col];
            float t = fmaxf(e0, e1);
#pragma unroll
            for (int mm = 8; mm >= 1; mm >>= 1) t = fmaxf(t, __shfl_xor(t, mm, 16));
            float mn = fmaxf(mrow[i], t);
            float fi = __expf(mrow[i] - mn);
            float p0 = __expf(e0 - mn);
            float p1 = __expf(e1 - mn);
            float rs = p0 + p1;
#pragma unroll
            for (int mm = 8; mm >= 1; mm >>= 1) rs += __shfl_xor(rs, mm, 16);
            srow[i] = srow[i] * fi + rs;
            mrow[i] = mn;
            f[i] = fi;
            int r = i + hf * 8;
            pl[r * 32 + col]      = (__bf16)p0;
            pl[r * 32 + 16 + col] = (__bf16)p1;
        }
#pragma unroll
        for (int i = 0; i < 8; ++i) {
            o0[i] *= f[i]; o1[i] *= f[i]; o2[i] *= f[i]; o3[i] *= f[i];
        }
        v16bf ap = load_a_bf16(pl, 32, 0);
        o0 = wmma_bf16(ap, load_b_bf16(vbase + (size_t)0 * 16 * SEQ, SEQ, k0), o0);
        o1 = wmma_bf16(ap, load_b_bf16(vbase + (size_t)1 * 16 * SEQ, SEQ, k0), o1);
        o2 = wmma_bf16(ap, load_b_bf16(vbase + (size_t)2 * 16 * SEQ, SEQ, k0), o2);
        o3 = wmma_bf16(ap, load_b_bf16(vbase + (size_t)3 * 16 * SEQ, SEQ, k0), o3);
    }
    float hm = hmask[h];
#pragma unroll
    for (int i = 0; i < 8; ++i) {
        int ll = qt * 16 + i + hf * 8;
        float inv = hm / srow[i];
        size_t base = ((size_t)b * SEQ + ll) * DM + (size_t)h * HDIM + col;
        O[base + 0]  = (__bf16)(o0[i] * inv);
        O[base + 16] = (__bf16)(o1[i] * inv);
        O[base + 32] = (__bf16)(o2[i] * inv);
        O[base + 48] = (__bf16)(o3[i] * inv);
    }
}

// ---- LayerNorm (one 256-thread block per token) ----
__global__ __launch_bounds__(256)
void ln_kernel(const float* __restrict__ X, const float* __restrict__ g,
               const float* __restrict__ bb, float* __restrict__ outF,
               __bf16* __restrict__ outB) {
    __shared__ float red[256];
    int t = blockIdx.x, tid = threadIdx.x;
    const float* x = X + (size_t)t * DM;
    float s = 0.f;
    for (int j = tid; j < DM; j += 256) s += x[j];
    red[tid] = s; __syncthreads();
    for (int st = 128; st > 0; st >>= 1) { if (tid < st) red[tid] += red[tid + st]; __syncthreads(); }
    float mean = red[0] * (1.f / DM); __syncthreads();
    float v = 0.f;
    for (int j = tid; j < DM; j += 256) { float d = x[j] - mean; v += d * d; }
    red[tid] = v; __syncthreads();
    for (int st = 128; st > 0; st >>= 1) { if (tid < st) red[tid] += red[tid + st]; __syncthreads(); }
    float rstd = rsqrtf(red[0] * (1.f / DM) + 1e-5f);
    for (int j = tid; j < DM; j += 256) {
        float y = (x[j] - mean) * rstd * g[j] + bb[j];
        outF[(size_t)t * DM + j] = y;
        outB[(size_t)t * DM + j] = (__bf16)y;
    }
}

// ---- Final: LN2 then scale by gate value, write d_out ----
__global__ __launch_bounds__(256)
void final_kernel(const float* __restrict__ Y, const float* __restrict__ g,
                  const float* __restrict__ bb, const float* __restrict__ gval,
                  float* __restrict__ out) {
    __shared__ float red[256];
    int t = blockIdx.x, tid = threadIdx.x;
    const float* x = Y + (size_t)t * DM;
    float s = 0.f;
    for (int j = tid; j < DM; j += 256) s += x[j];
    red[tid] = s; __syncthreads();
    for (int st = 128; st > 0; st >>= 1) { if (tid < st) red[tid] += red[tid + st]; __syncthreads(); }
    float mean = red[0] * (1.f / DM); __syncthreads();
    float v = 0.f;
    for (int j = tid; j < DM; j += 256) { float d = x[j] - mean; v += d * d; }
    red[tid] = v; __syncthreads();
    for (int st = 128; st > 0; st >>= 1) { if (tid < st) red[tid] += red[tid + st]; __syncthreads(); }
    float rstd = rsqrtf(red[0] * (1.f / DM) + 1e-5f);
    float gw = gval[t];
    for (int j = tid; j < DM; j += 256) {
        float y = (x[j] - mean) * rstd * g[j] + bb[j];
        out[(size_t)t * DM + j] = gw * y;
    }
}

// ---- Gate: one wave per token; logits, softmax-top1, bucket counts ----
__global__ __launch_bounds__(128)
void gate_kernel(const float* __restrict__ H, const float* __restrict__ Wg,
                 const float* __restrict__ bg, const int* __restrict__ idxes,
                 float* __restrict__ gval, int* __restrict__ gidx, int* __restrict__ cnt) {
    int wave = threadIdx.x >> 5, lane = threadIdx.x & 31;
    int t = blockIdx.x * 4 + wave;
    if (t >= TKN) return;
    int b = t >> 10;
    int ds = idxes[b];
    const float* x = H + (size_t)t * DM;
    const float* W = Wg + (size_t)ds * NEXP * DM;
    float a0 = 0.f, a1 = 0.f, a2 = 0.f, a3 = 0.f;
    for (int j = lane; j < DM; j += 32) {
        float xv = x[j];
        a0 += xv * W[j];
        a1 += xv * W[DM + j];
        a2 += xv * W[2 * DM + j];
        a3 += xv * W[3 * DM + j];
    }
#pragma unroll
    for (int mm = 16; mm >= 1; mm >>= 1) {
        a0 += __shfl_xor(a0, mm, 32); a1 += __shfl_xor(a1, mm, 32);
        a2 += __shfl_xor(a2, mm, 32); a3 += __shfl_xor(a3, mm, 32);
    }
    if (lane == 0) {
        float lg[4] = { a0 + bg[ds * NEXP + 0], a1 + bg[ds * NEXP + 1],
                        a2 + bg[ds * NEXP + 2], a3 + bg[ds * NEXP + 3] };
        int am = 0; float lm = lg[0];
        if (lg[1] > lm) { lm = lg[1]; am = 1; }
        if (lg[2] > lm) { lm = lg[2]; am = 2; }
        if (lg[3] > lm) { lm = lg[3]; am = 3; }
        float sum = __expf(lg[0] - lm) + __expf(lg[1] - lm) +
                    __expf(lg[2] - lm) + __expf(lg[3] - lm);
        gval[t] = 1.0f / sum;
        gidx[t] = am;
        atomicAdd(&cnt[am], 1);
    }
}

__global__ void route_init(int* cnt, int* cursor) {
    if (threadIdx.x < NEXP) { cnt[threadIdx.x] = 0; cursor[threadIdx.x] = 0; }
}
__global__ void route_offsets(const int* cnt, int* eoff, int* cursor) {
    if (threadIdx.x == 0) {
        int a = 0;
        for (int e = 0; e < NEXP; ++e) { eoff[e] = a; cursor[e] = a; a += cnt[e]; }
    }
}
__global__ __launch_bounds__(256)
void route_scatter(const int* __restrict__ gidx, int* cursor, int* __restrict__ perm) {
    int t = blockIdx.x * 256 + threadIdx.x;
    if (t < TKN) {
        int e = gidx[t];
        int p = atomicAdd(&cursor[e], 1);
        perm[p] = t;
    }
}

// ---- Routed expert fc1: gathered rows -> gelu -> compact HH[pos, INTER] ----
__global__ __launch_bounds__(128)
void moe_fc1(const __bf16* __restrict__ Xb, const float* __restrict__ W1,
             const float* __restrict__ B1, const int* __restrict__ perm,
             const int* __restrict__ cnt, const int* __restrict__ eoff,
             __bf16* __restrict__ HH) {
    int wave = threadIdx.x >> 5, lane = threadIdx.x & 31;
    unsigned id = blockIdx.x * 4u + wave;     // e*65536 + tm*256 + tn
    int e = id >> 16;
    unsigned rem = id & 0xFFFFu;
    int tm = rem >> 8, tn = rem & 255;
    int ce = cnt[e];
    if (tm * 16 >= ce) return;
    int r = lane & 15, hf = lane >> 4, col = r;
    int m_ln = tm * 16 + r;
    int pidx = eoff[e] + ((m_ln < ce) ? m_ln : (ce - 1));
    int tok = perm[pidx];
    const __bf16* arow = Xb + (size_t)tok * DM;
    const float* Wp = W1 + ((size_t)e * INTER_ + (size_t)tn * 16) * DM;
    v8f acc = V8F_ZERO;
    for (int k = 0; k < DM; k += 32)
        acc = wmma_bf16(load_a_row(arow, k), load_b_f32(Wp, DM, k), acc);
    float bs = B1[e * INTER_ + tn * 16 + col];
#pragma unroll
    for (int i = 0; i < 8; ++i) {
        int m = tm * 16 + i + hf * 8;
        if (m < ce) {
            float gv = gelu_exact(acc[i] + bs);
            HH[((size_t)(eoff[e] + m)) * INTER_ + tn * 16 + col] = (__bf16)gv;
        }
    }
}

// ---- Routed expert fc2: HH @ W2^T, scatter-add into ysh[token, :] ----
__global__ __launch_bounds__(128)
void moe_fc2(const __bf16* __restrict__ HH, const float* __restrict__ W2,
             const int* __restrict__ perm, const int* __restrict__ cnt,
             const int* __restrict__ eoff, float* __restrict__ Y) {
    int wave = threadIdx.x >> 5, lane = threadIdx.x & 31;
    unsigned id = blockIdx.x * 4u + wave;     // e*16384 + tm*64 + tn
    int e = id >> 14;
    unsigned rem = id & 0x3FFFu;
    int tm = rem >> 6, tn = rem & 63;
    int ce = cnt[e];
    if (tm * 16 >= ce) return;
    int r = lane & 15, hf = lane >> 4, col = r;
    int m_ln = tm * 16 + r;
    int p = eoff[e] + ((m_ln < ce) ? m_ln : (ce - 1));
    const __bf16* arow = HH + (size_t)p * INTER_;
    const float* Wp = W2 + ((size_t)e * DM + (size_t)tn * 16) * INTER_;
    v8f acc = V8F_ZERO;
    for (int k = 0; k < INTER_; k += 32)
        acc = wmma_bf16(load_a_row(arow, k), load_b_f32(Wp, INTER_, k), acc);
#pragma unroll
    for (int i = 0; i < 8; ++i) {
        int m = tm * 16 + i + hf * 8;
        if (m < ce) {
            int tok = perm[eoff[e] + m];
            Y[(size_t)tok * DM + tn * 16 + col] += acc[i];  // unique writer per element
        }
    }
}

extern "C" void kernel_launch(void* const* d_in, const int* in_sizes, int n_in,
                              void* d_out, int out_size, void* d_ws, size_t ws_size,
                              hipStream_t stream) {
    (void)in_sizes; (void)n_in; (void)out_size; (void)ws_size;
    const float* hs     = (const float*)d_in[0];
    const float* mask   = (const float*)d_in[1];
    const float* hmask  = (const float*)d_in[2];
    const int*   idxes  = (const int*)d_in[3];
    const float* q_w    = (const float*)d_in[4];
    const float* k_w    = (const float*)d_in[5];
    const float* v_w    = (const float*)d_in[6];
    const float* o_w    = (const float*)d_in[7];
    const float* fc1_w  = (const float*)d_in[8];
    const float* fc2_w  = (const float*)d_in[9];
    const float* efc1_w = (const float*)d_in[10];
    const float* efc2_w = (const float*)d_in[11];
    const float* gate_w = (const float*)d_in[12];
    const float* q_b    = (const float*)d_in[13];
    const float* k_b    = (const float*)d_in[14];
    const float* v_b    = (const float*)d_in[15];
    const float* o_b    = (const float*)d_in[16];
    const float* fc1_b  = (const float*)d_in[17];
    const float* fc2_b  = (const float*)d_in[18];
    const float* efc1_b = (const float*)d_in[19];
    const float* gate_b = (const float*)d_in[20];
    const float* ln1_g  = (const float*)d_in[21];
    const float* ln1_b  = (const float*)d_in[22];
    const float* ln2_g  = (const float*)d_in[23];
    const float* ln2_b  = (const float*)d_in[24];

    char* ws = (char*)d_ws;
    size_t off = 0;
    auto take = [&](size_t n) -> char* {
        char* p = ws + off;
        off += (n + 255) & ~(size_t)255;
        return p;
    };
    __bf16* x_bf  = (__bf16*)take((size_t)TKN * DM * 2);
    __bf16* qs    = (__bf16*)take((size_t)TKN * DM * 2);
    __bf16* ks    = (__bf16*)take((size_t)TKN * DM * 2);
    __bf16* vt    = (__bf16*)take((size_t)TKN * DM * 2);
    __bf16* attnb = (__bf16*)take((size_t)TKN * DM * 2);
    float*  h1    = (float*) take((size_t)TKN * DM * 4);
    float*  h     = (float*) take((size_t)TKN * DM * 4);
    __bf16* h_bf  = (__bf16*)take((size_t)TKN * DM * 2);
    float*  ysh   = (float*) take((size_t)TKN * DM * 4);
    __bf16* hh_sh = (__bf16*)take((size_t)TKN * FFN_ * 2);
    __bf16* hh_ex = (__bf16*)take((size_t)TKN * INTER_ * 2);
    float*  gval  = (float*) take((size_t)TKN * 4);
    int*    gidx  = (int*)   take((size_t)TKN * 4);
    int*    perm  = (int*)   take((size_t)TKN * 4);
    int*    cnt   = (int*)   take(64);
    int*    eoff  = (int*)   take(64);
    int*    curs  = (int*)   take(64);

    dim3 blk(128);
    const int gProj = (TKN / 64) * (DM / 16);    // 4096 blocks (4 M-tiles/block)
    const int gFfn  = (TKN / 64) * (FFN_ / 16);  // 16384 blocks
    const float qscale = 0.125f;                 // HD^-0.5

    // hidden -> bf16
    cvt_kernel<<<(TKN * DM / 4 + 255) / 256, 256, 0, stream>>>(hs, x_bf, TKN * DM / 4);

    // Q (scaled, [B,NH,L,HD]), K ([B,NH,L,HD]), V^T ([B,NH,HD,L])
    gemm_bf16w<2><<<gProj, blk, 0, stream>>>(x_bf, q_w, q_b, nullptr, nullptr, qs, TKN, DM, DM, qscale);
    gemm_bf16w<2><<<gProj, blk, 0, stream>>>(x_bf, k_w, k_b, nullptr, nullptr, ks, TKN, DM, DM, 1.0f);
    gemm_bf16w<3><<<gProj, blk, 0, stream>>>(x_bf, v_w, v_b, nullptr, nullptr, vt, TKN, DM, DM, 1.0f);

    // attention -> attnb (bf16, token-major)
    attn_kernel<<<(BATCH * NHEAD * (SEQ / 16)) / 4, blk, 0, stream>>>(qs, ks, vt, mask, hmask, attnb);

    // o-proj + residual -> h1; LN1 -> h (f32) + h_bf
    gemm_bf16w<5><<<gProj, blk, 0, stream>>>(attnb, o_w, o_b, hs, h1, nullptr, TKN, DM, DM, 1.0f);
    ln_kernel<<<TKN, 256, 0, stream>>>(h1, ln1_g, ln1_b, h, h_bf);

    // shared FFN (computed once): gelu(h@fc1^T+b1) -> hh_sh; hh_sh@fc2^T+b2 + h -> ysh
    gemm_bf16w<4><<<gFfn, blk, 0, stream>>>(h_bf, fc1_w, fc1_b, nullptr, nullptr, hh_sh, TKN, FFN_, DM, 1.0f);
    gemm_bf16w<5><<<gProj, blk, 0, stream>>>(hh_sh, fc2_w, fc2_b, h, ysh, nullptr, TKN, DM, FFN_, 1.0f);

    // top-1 routing
    route_init<<<1, 32, 0, stream>>>(cnt, curs);
    gate_kernel<<<TKN / 4, blk, 0, stream>>>(h, gate_w, gate_b, idxes, gval, gidx, cnt);
    route_offsets<<<1, 32, 0, stream>>>(cnt, eoff, curs);
    route_scatter<<<TKN / 256, 256, 0, stream>>>(gidx, curs, perm);

    // routed expert FFN (each token only through its expert)
    moe_fc1<<<(NEXP * (TKN / 16) * (INTER_ / 16)) / 4, blk, 0, stream>>>(
        h_bf, efc1_w, efc1_b, perm, cnt, eoff, hh_ex);
    moe_fc2<<<(NEXP * (TKN / 16) * (DM / 16)) / 4, blk, 0, stream>>>(
        hh_ex, efc2_w, perm, cnt, eoff, ysh);

    // LN2 + gate scaling -> d_out (fp32)
    final_kernel<<<TKN, 256, 0, stream>>>(ysh, ln2_g, ln2_b, gval, (float*)d_out);
}
